// DistanceWeightedLoss_50362786512919
// MI455X (gfx1250) — compile-verified, matrix operand order checked by reading, not run on hardware
//
#include <hip/hip_runtime.h>
#include <hip/hip_bf16.h>
#include <math.h>

#define N 4096
#define CC 21
#define LD (N + CC)
#define MARGIN 0.09f
#define CUTOFF (-0.03f)
#define NZCUT 0.09f
#define COEFA (-4094.0f)   /* 2 - d,      d = 4096 */
#define COEFB (-2046.5f)   /* -(d-3)/2               */

typedef __attribute__((ext_vector_type(16))) __bf16 v16bf;
typedef __attribute__((ext_vector_type(8)))  float  v8f;

__device__ __forceinline__ float logw_of(float st) {
    return COEFA * __logf(st) + COEFB * __logf(1.0f - 0.25f * st * st);
}

__device__ __forceinline__ unsigned hash3(unsigned a, unsigned b, unsigned c) {
    unsigned h = a * 0x8DA6B343u ^ b * 0xD8163841u ^ c * 0xCB1AB31Fu;
    h += 0x9E3779B9u;
    h ^= h >> 15; h *= 0x2C1B3C6Du;
    h ^= h >> 12; h *= 0x297A2D39u;
    h ^= h >> 15;
    return h;
}

__device__ __forceinline__ float u01(unsigned h) {
    return (h >> 8) * 0x1p-24f + 0x1p-25f;   /* strictly in (0,1) */
}

__global__ void k_gt(const float* __restrict__ x, float* __restrict__ gt) {
    int i = blockIdx.x * blockDim.x + threadIdx.x;
    if (i < N) gt[i] = 1.0f - x[(size_t)i * LD];
}

/* Pass 1: per-row partial max of log_w for both orientations.
 * Element scores[i][j] contributes lw0 with gt[i] (row-major matrix)
 * and lw1 with gt[j] (the transposed matrix at [j][i]).               */
__global__ __launch_bounds__(256)
void k_rowmax(const float* __restrict__ x, const float* __restrict__ gt,
              float* __restrict__ pm0, float* __restrict__ pm1) {
    __shared__ float r0[256], r1[256];
    const int i = blockIdx.x, tid = threadIdx.x;
    const float LWTINY = logw_of(1e-10f);
    const float gi = gt[i];
    const float* row = x + (size_t)i * LD + CC;
    float m0 = -3.4e38f, m1 = -3.4e38f;
    for (int j = tid; j < N; j += 256) {
        __builtin_prefetch(row + j + 512, 0, 0);     /* global_prefetch_b8 */
        float s  = 1.0f - row[j];
        float lw = logw_of(s);
        float l0 = (s - gi    < CUTOFF) ? LWTINY : lw;
        float l1 = (s - gt[j] < CUTOFF) ? LWTINY : lw;
        m0 = fmaxf(m0, l0);
        m1 = fmaxf(m1, l1);
    }
    r0[tid] = m0; r1[tid] = m1; __syncthreads();
    for (int st = 128; st > 0; st >>= 1) {
        if (tid < st) {
            r0[tid] = fmaxf(r0[tid], r0[tid + st]);
            r1[tid] = fmaxf(r1[tid], r1[tid + st]);
        }
        __syncthreads();
    }
    if (tid == 0) { pm0[i] = r0[0]; pm1[i] = r1[0]; }
}

__global__ __launch_bounds__(256)
void k_gmax(const float* __restrict__ pm0, const float* __restrict__ pm1,
            float* __restrict__ gmax) {
    __shared__ float r0[256], r1[256];
    const int tid = threadIdx.x;
    float m0 = -3.4e38f, m1 = -3.4e38f;
    for (int j = tid; j < N; j += 256) {
        m0 = fmaxf(m0, pm0[j]);
        m1 = fmaxf(m1, pm1[j]);
    }
    r0[tid] = m0; r1[tid] = m1; __syncthreads();
    for (int st = 128; st > 0; st >>= 1) {
        if (tid < st) {
            r0[tid] = fmaxf(r0[tid], r0[tid + st]);
            r1[tid] = fmaxf(r1[tid], r1[tid + st]);
        }
        __syncthreads();
    }
    if (tid == 0) { gmax[0] = r0[0]; gmax[1] = r1[0]; }
}

/* Pass 2: one 256-thread block per row.
 *  - async-DMA the row/column of x into LDS (global_load_async_to_lds_b32;
 *    per-lane addresses handle the strided transposed gather too)
 *  - build weights, sum them with one v_wmma_f32_16x16x32_bf16 per wave
 *  - inclusive prefix scan of p_j = w_j/Sden + 1e-30 in LDS
 *  - k inverse-CDF samples via binary search + hinge accumulation       */
__global__ __launch_bounds__(256)
void k_sample(const float* __restrict__ x, const float* __restrict__ gt,
              const float* __restrict__ gmax, int orient, int ksamp,
              float* __restrict__ rowloss) {
    __shared__ float wlds[N];     /* raw x -> weights -> prefix sums */
    __shared__ float slds[N];     /* distances s                      */
    __shared__ float red[256];
    __shared__ float wsum[8];
    __shared__ float sred[2];
    __shared__ float hred[16];
    const int i = blockIdx.x, tid = threadIdx.x;
    const float gi = gt[i];
    const float gm = gmax[orient];
    const float LWTINY = logw_of(1e-10f);

    /* Stage raw x values into LDS via async loads (ASYNCcnt-tracked DMA). */
    for (int j = tid; j < N; j += 256) {
        size_t off = (orient == 0) ? (size_t)i * LD + CC + j
                                   : (size_t)j * LD + CC + i;
        unsigned long long gaddr = (unsigned long long)(uintptr_t)(x + off);
        unsigned ldsoff = (unsigned)(uintptr_t)&wlds[j];   /* LDS byte offset */
        asm volatile("global_load_async_to_lds_b32 %0, %1, off"
                     :: "v"(ldsoff), "v"(gaddr) : "memory");
    }
    asm volatile("s_wait_asynccnt 0x0" ::: "memory");
    /* Each thread transforms only the slots it issued itself. */
    for (int j = tid; j < N; j += 256) {
        float s = 1.0f - wlds[j];
        slds[j] = s;
        float diff = s - gi;
        float lw = (diff < CUTOFF) ? LWTINY : logw_of(s);
        float w  = __expf(lw - gm);
        if (j == i) w = 0.0f;
        if (!(diff > NZCUT)) w *= 1e-10f;
        wlds[j] = w;
    }
    __syncthreads();

    /* WMMA chunk sums: wave wv owns wlds[wv*512 .. wv*512+512) */
    const int wv = tid >> 5, lane = tid & 31;
    v16bf a, ones;
#pragma unroll
    for (int t = 0; t < 16; ++t) ones[t] = (__bf16)1.0f;
    const int base = wv * 512 + lane * 16;
#pragma unroll
    for (int t = 0; t < 16; ++t) a[t] = (__bf16)wlds[base + t];
    v8f acc = {};
    acc = __builtin_amdgcn_wmma_f32_16x16x32_bf16(
        false, a, false, ones, (short)0, acc, false, false);
    float cs = acc[0] + acc[1] + acc[2] + acc[3] +
               acc[4] + acc[5] + acc[6] + acc[7];     /* half-tile sum */
    float other = __shfl(cs, lane ^ 16, 32);
    if (lane == 0) wsum[wv] = cs + other;             /* full 512-chunk sum */
    __syncthreads();
    if (tid == 0) {
        float S = 0.0f;
        for (int t = 0; t < 8; ++t) S += wsum[t];
        sred[1] = S + 1e-10f;                         /* Sden */
    }
    __syncthreads();
    const float Sden = sred[1];

    /* Inclusive prefix scan of p_j = w_j/Sden + 1e-30 (inverse-CDF table).
     * Thread tid owns contiguous segment [tid*16, tid*16+16).            */
    {
        const int b = tid * 16;
        float run = 0.0f;
#pragma unroll
        for (int e = 0; e < 16; ++e) {
            run += wlds[b + e] / Sden + 1e-30f;
            wlds[b + e] = run;
        }
        red[tid] = run;
    }
    __syncthreads();
    if (tid == 0) {                                   /* scan 256 segment totals */
        float acc2 = 0.0f;
        for (int t = 0; t < 256; ++t) { float v = red[t]; red[t] = acc2; acc2 += v; }
        sred[0] = acc2;                               /* grand total */
    }
    __syncthreads();
    {
        const float offp = red[tid];
        const int b = tid * 16;
#pragma unroll
        for (int e = 0; e < 16; ++e) wlds[b + e] += offp;
    }
    __syncthreads();

    /* k categorical samples: u ~ U(0,total), binary search the CDF. */
    const float total = sred[0];
    const float diag = slds[i];
    if (tid < ksamp) {
        float u = u01(hash3((unsigned)((i << 1) | orient), (unsigned)tid,
                            0x51ED270Bu)) * total;
        int lo = 0, hi = N - 1;
        while (lo < hi) {
            int mid = (lo + hi) >> 1;
            if (wlds[mid] >= u) hi = mid; else lo = mid + 1;
        }
        hred[tid] = fmaxf(0.0f, MARGIN + diag - slds[lo]);
    }
    __syncthreads();
    if (tid == 0) {
        float hs = 0.0f;
        for (int t = 0; t < ksamp; ++t) hs += hred[t];
        rowloss[i] = 1.0f + hs;
    }
}

__global__ __launch_bounds__(256)
void k_final(const float* __restrict__ rl0, const float* __restrict__ rl1,
             float* __restrict__ out) {
    __shared__ float r[256];
    const int tid = threadIdx.x;
    float s = 0.0f;
    for (int j = tid; j < N; j += 256) s += rl0[j] + rl1[j];
    r[tid] = s; __syncthreads();
    for (int st = 128; st > 0; st >>= 1) {
        if (tid < st) r[tid] += r[tid + st];
        __syncthreads();
    }
    if (tid == 0) out[0] = r[0] / (float)N;   /* mean(loss0) + mean(loss1) */
}

extern "C" void kernel_launch(void* const* d_in, const int* in_sizes, int n_in,
                              void* d_out, int out_size, void* d_ws, size_t ws_size,
                              hipStream_t stream) {
    const float* x = (const float*)d_in[0];   /* (4096, 4117) f32; labels unused */
    float* ws   = (float*)d_ws;
    float* gt   = ws;                 /* N      */
    float* pm0  = ws + N;             /* N      */
    float* pm1  = ws + 2 * N;         /* N      */
    float* gmax = ws + 3 * N;         /* 2 (+pad) */
    float* rl0  = ws + 3 * N + 64;    /* N      */
    float* rl1  = rl0 + N;            /* N      */

    k_gt<<<N / 256, 256, 0, stream>>>(x, gt);
    k_rowmax<<<N, 256, 0, stream>>>(x, gt, pm0, pm1);
    k_gmax<<<1, 256, 0, stream>>>(pm0, pm1, gmax);
    k_sample<<<N, 256, 0, stream>>>(x, gt, gmax, 0, 5,  rl0);
    k_sample<<<N, 256, 0, stream>>>(x, gt, gmax, 1, 10, rl1);
    k_final<<<1, 256, 0, stream>>>(rl0, rl1, (float*)d_out);
}
